// LoRACAAttentionProcessor_10514079941019
// MI455X (gfx1250) — matrix operand order repaired and checked by reading, not compiled, not run on hardware
//
#include <hip/hip_runtime.h>

typedef _Float16 h4_t  __attribute__((ext_vector_type(4)));
typedef _Float16 h8_t  __attribute__((ext_vector_type(8)));
typedef _Float16 h16_t __attribute__((ext_vector_type(16)));
typedef float    f4_t  __attribute__((ext_vector_type(4)));
typedef float    f8_t  __attribute__((ext_vector_type(8)));
typedef int      i4_t  __attribute__((ext_vector_type(4)));

typedef __attribute__((address_space(1))) i4_t* gvec_p;  // global i32x4*
typedef __attribute__((address_space(3))) i4_t* lvec_p;  // LDS i32x4*

#define WMMA16(a, b, c) __builtin_amdgcn_wmma_f32_16x16x32_f16( \
    false, (a), false, (b), (short)0, (c), false, false)

#ifndef __has_builtin
#define __has_builtin(x) 0
#endif
#if __has_builtin(__builtin_amdgcn_global_load_async_to_lds_b128)
#define ASYNC_CP 1
#else
#define ASYNC_CP 0
#endif

#if ASYNC_CP && __has_builtin(__builtin_amdgcn_s_wait_asynccnt)
#define WAIT_ASYNC(n) __builtin_amdgcn_s_wait_asynccnt(n)
#elif ASYNC_CP
#define WAIT_ASYNC(n) asm volatile("s_wait_asynccnt %0" ::"i"(n) : "memory")
#else
#define WAIT_ASYNC(n) ((void)0)
#endif

__device__ __forceinline__ void cp16(const _Float16* g, _Float16* l) {
#if ASYNC_CP
  __builtin_amdgcn_global_load_async_to_lds_b128(
      (gvec_p)(void*)const_cast<_Float16*>(g), (lvec_p)(void*)l, 0, 0);
#else
  *(h8_t*)l = *(const h8_t*)g;
#endif
}

__device__ inline f8_t f8zero() {
  f8_t z;
#pragma unroll
  for (int q = 0; q < 8; ++q) z[q] = 0.0f;
  return z;
}
__device__ inline h8_t h8zero() {
  h8_t z;
#pragma unroll
  for (int q = 0; q < 8; ++q) z[q] = (_Float16)0.0f;
  return z;
}

// ---------------------------------------------------------------- cvt f32->f16
__global__ __launch_bounds__(256) void cvt_f32_f16(
    const float* __restrict__ in, _Float16* __restrict__ out, int n4) {
  int i = blockIdx.x * blockDim.x + threadIdx.x;
  if (i >= n4) return;
  f4_t v = *(const f4_t*)(in + (size_t)i * 4);
  h4_t h = {(_Float16)v.x, (_Float16)v.y, (_Float16)v.z, (_Float16)v.w};
  *(h4_t*)(out + (size_t)i * 4) = h;
}

// ----------------------------------------------- fold LoRA into base weight
// outT[n][k] = f16( W[k][n] + sum_r down[k][r] * up[r][n] )  (TRANSPOSED out)
__global__ __launch_bounds__(256) void fuse_lora_wT(
    const float* __restrict__ W, const float* __restrict__ down,
    const float* __restrict__ up, _Float16* __restrict__ outT, int N, int K) {
  int n = blockIdx.x * blockDim.x + threadIdx.x;
  int k = blockIdx.y;
  if (n >= N) return;
  float acc = W[(size_t)k * N + n];
  const float* dk = down + (size_t)k * 64;
#pragma unroll 8
  for (int r = 0; r < 64; ++r) acc += dk[r] * up[(size_t)r * N + n];
  outT[(size_t)n * K + k] = (_Float16)acc;
}

// ---------------------------------------------------------------- WMMA GEMM
// A [M,K] f16 row-major, Bt [N,K] f16 (pre-transposed). Block tile 128x128,
// BK=64, async double-buffered LDS. 8 waves; each wave 32(M) x 64(N).
template <bool OUT_F32>
__global__ __launch_bounds__(256) void gemm_f16t(
    const _Float16* __restrict__ A, const _Float16* __restrict__ Bt,
    void* __restrict__ Cout, const float* __restrict__ bias,
    int M, int N, int K) {
  __shared__ _Float16 sA[2][128][64];
  __shared__ _Float16 sB[2][128][64];  // [n][k]

  const int tid = threadIdx.x;
  const int lane = tid & 31, wid = tid >> 5;
  const int lh = (lane >> 4) & 1, l16 = lane & 15;
  const int bm = blockIdx.y * 128, bn = blockIdx.x * 128;
  const int wm = (wid >> 1) * 32, wn = (wid & 1) * 64;

  // staging: each thread copies half a row (32 halves = 64B) of A and B tiles
  const int srow = tid >> 1;
  const int scol = (tid & 1) * 32;
  const int arow = (bm + srow < M) ? (bm + srow) : (M - 1);  // clamp; junk rows discarded at store
  const _Float16* gA = A + (size_t)arow * K + scol;
  const _Float16* gB = Bt + (size_t)(bn + srow) * K + scol;

  const int T = K >> 6;

  auto stage = [&](int t, int buf) {
    const _Float16* pa = gA + (size_t)t * 64;
    const _Float16* pb = gB + (size_t)t * 64;
    _Float16* la = &sA[buf][srow][scol];
    _Float16* lb = &sB[buf][srow][scol];
#pragma unroll
    for (int i = 0; i < 4; ++i) cp16(pa + i * 8, la + i * 8);
#pragma unroll
    for (int i = 0; i < 4; ++i) cp16(pb + i * 8, lb + i * 8);
  };

  f8_t acc[2][4];
#pragma unroll
  for (int i = 0; i < 2; ++i)
#pragma unroll
    for (int j = 0; j < 4; ++j) acc[i][j] = f8zero();

  stage(0, 0);
  if (T > 1) stage(1, 1);

  for (int t = 0; t < T; ++t) {
    if (t + 1 < T) WAIT_ASYNC(8);  // keep next stage (8 ops) in flight
    else           WAIT_ASYNC(0);
    __syncthreads();
    const int buf = t & 1;
#pragma unroll
    for (int ks = 0; ks < 2; ++ks) {
      h16_t af[2], bf[4];
#pragma unroll
      for (int i = 0; i < 2; ++i) {
        int r = wm + i * 16 + l16;
        int k0 = ks * 32 + lh * 8;  // A layout: chunks {k0..k0+7, k0+16..k0+23}
        ((h8_t*)&af[i])[0] = *(const h8_t*)&sA[buf][r][k0];
        ((h8_t*)&af[i])[1] = *(const h8_t*)&sA[buf][r][k0 + 16];
      }
#pragma unroll
      for (int j = 0; j < 4; ++j) {
        int n = wn + j * 16 + l16;
        int k0 = ks * 32 + lh * 8;  // B layout: chunks {k0..k0+7, k0+16..k0+23}
        ((h8_t*)&bf[j])[0] = *(const h8_t*)&sB[buf][n][k0];
        ((h8_t*)&bf[j])[1] = *(const h8_t*)&sB[buf][n][k0 + 16];
      }
#pragma unroll
      for (int i = 0; i < 2; ++i)
#pragma unroll
        for (int j = 0; j < 4; ++j) acc[i][j] = WMMA16(af[i], bf[j], acc[i][j]);
    }
    __syncthreads();
    if (t + 2 < T) stage(t + 2, buf);
  }

#pragma unroll
  for (int i = 0; i < 2; ++i)
#pragma unroll
    for (int j = 0; j < 4; ++j)
#pragma unroll
      for (int r = 0; r < 8; ++r) {
        int row = bm + wm + i * 16 + r + lh * 8;
        int col = bn + wn + j * 16 + l16;
        if (row < M) {
          float v = acc[i][j][r];
          if (OUT_F32)
            ((float*)Cout)[(size_t)row * N + col] = v + (bias ? bias[col] : 0.0f);
          else
            ((_Float16*)Cout)[(size_t)row * N + col] = (_Float16)v;
        }
      }
}

// ---------------------------------------------------------------- attention
// Q [b*4096,1280] f16, K/V [b*77,1280] f16 (head h at cols h*64..).
// grid: (32 q-tiles, 320 b*h). Each wave handles 16 q rows; skv padded to 96.
__global__ __launch_bounds__(256) void attn_kernel(
    const _Float16* __restrict__ Q, const _Float16* __restrict__ Kp,
    const _Float16* __restrict__ Vp, _Float16* __restrict__ Out) {
  __shared__ _Float16 sK[96][64];      // [key][d]
  __shared__ _Float16 sV[64][96];      // [d][key]  (transposed)
  __shared__ _Float16 sP[8][16][96];   // per-wave probs

  const int bh = blockIdx.y, b = bh / 20, h = bh % 20;
  const int qt = blockIdx.x;
  const int tid = threadIdx.x, lane = tid & 31, wid = tid >> 5;
  const int lh = (lane >> 4) & 1, l16 = lane & 15;

  if (tid < 96) {
    int key = tid;
    if (key < 77) {
      const _Float16* pk = Kp + ((size_t)(b * 77 + key)) * 1280 + h * 64;
      const _Float16* pv = Vp + ((size_t)(b * 77 + key)) * 1280 + h * 64;
#pragma unroll
      for (int c = 0; c < 8; ++c) *(h8_t*)&sK[key][c * 8] = *(const h8_t*)(pk + c * 8);
#pragma unroll
      for (int d = 0; d < 64; ++d) sV[d][key] = pv[d];
    } else {
      h8_t z = h8zero();
#pragma unroll
      for (int c = 0; c < 8; ++c) *(h8_t*)&sK[key][c * 8] = z;
#pragma unroll
      for (int d = 0; d < 64; ++d) sV[d][key] = (_Float16)0.0f;
    }
  }
  __syncthreads();

  const int qrow0 = qt * 128 + wid * 16;
  const size_t qbase = (size_t)b * 4096 + qrow0;

  h16_t aq[2];
#pragma unroll
  for (int i = 0; i < 2; ++i) {
    const _Float16* p = Q + (qbase + l16) * 1280 + h * 64 + i * 32 + lh * 8;
    ((h8_t*)&aq[i])[0] = *(const h8_t*)p;
    ((h8_t*)&aq[i])[1] = *(const h8_t*)(p + 16);
  }

  // scores = Q @ K^T  (5 N-tiles of 16 keys, d=64 -> 2 K-steps)
  f8_t s[5];
#pragma unroll
  for (int nt = 0; nt < 5; ++nt) {
    s[nt] = f8zero();
#pragma unroll
    for (int i = 0; i < 2; ++i) {
      h16_t bk;
      ((h8_t*)&bk)[0] = *(const h8_t*)&sK[nt * 16 + l16][i * 32 + lh * 8];
      ((h8_t*)&bk)[1] = *(const h8_t*)&sK[nt * 16 + l16][i * 32 + lh * 8 + 16];
      s[nt] = WMMA16(aq[i], bk, s[nt]);
    }
  }

  // softmax over 77 keys, wave-local shuffle reductions within 16-lane halves
  const float scale = 0.125f;  // 1/sqrt(64)
#pragma unroll
  for (int r = 0; r < 8; ++r) {
    float v[5];
    float rmax = -1e30f;
#pragma unroll
    for (int nt = 0; nt < 5; ++nt) {
      int col = nt * 16 + l16;
      v[nt] = (col < 77) ? s[nt][r] * scale : -1e30f;
      rmax = fmaxf(rmax, v[nt]);
    }
#pragma unroll
    for (int m = 1; m < 16; m <<= 1) rmax = fmaxf(rmax, __shfl_xor(rmax, m, 32));
    float e[5], rsum = 0.0f;
#pragma unroll
    for (int nt = 0; nt < 5; ++nt) {
      int col = nt * 16 + l16;
      e[nt] = (col < 77) ? __expf(v[nt] - rmax) : 0.0f;
      rsum += e[nt];
    }
#pragma unroll
    for (int m = 1; m < 16; m <<= 1) rsum += __shfl_xor(rsum, m, 32);
    float inv = 1.0f / rsum;
    int prow = r + lh * 8;
#pragma unroll
    for (int nt = 0; nt < 5; ++nt)
      sP[wid][prow][nt * 16 + l16] = (_Float16)(e[nt] * inv);
  }
  if (lane < 16) {  // zero pad keys 80..95
    h8_t z = h8zero();
    *(h8_t*)&sP[wid][l16][80] = z;
    *(h8_t*)&sP[wid][l16][88] = z;
  }
  // per-wave LDS region: DS ops are in-order within a wave, no barrier needed

  // out = P @ V   (K-dim 96 -> 3 steps, 4 N-tiles of d)
  f8_t o[4];
#pragma unroll
  for (int nt = 0; nt < 4; ++nt) o[nt] = f8zero();
#pragma unroll
  for (int kt = 0; kt < 3; ++kt) {
    h16_t ap;
    ((h8_t*)&ap)[0] = *(const h8_t*)&sP[wid][l16][kt * 32 + lh * 8];
    ((h8_t*)&ap)[1] = *(const h8_t*)&sP[wid][l16][kt * 32 + lh * 8 + 16];
#pragma unroll
    for (int nt = 0; nt < 4; ++nt) {
      h16_t bv;
      ((h8_t*)&bv)[0] = *(const h8_t*)&sV[nt * 16 + l16][kt * 32 + lh * 8];
      ((h8_t*)&bv)[1] = *(const h8_t*)&sV[nt * 16 + l16][kt * 32 + lh * 8 + 16];
      o[nt] = WMMA16(ap, bv, o[nt]);
    }
  }

#pragma unroll
  for (int nt = 0; nt < 4; ++nt)
#pragma unroll
    for (int r = 0; r < 8; ++r) {
      size_t row = qbase + r + lh * 8;
      int col = h * 64 + nt * 16 + l16;
      Out[row * 1280 + col] = (_Float16)o[nt][r];
    }
}

// ---------------------------------------------------------------- launcher
extern "C" void kernel_launch(void* const* d_in, const int* in_sizes, int n_in,
                              void* d_out, int out_size, void* d_ws, size_t ws_size,
                              hipStream_t stream) {
  const float* hs  = (const float*)d_in[0];
  const float* ehs = (const float*)d_in[1];
  const float* Wq  = (const float*)d_in[2];
  const float* Wk  = (const float*)d_in[3];
  const float* Wv  = (const float*)d_in[4];
  const float* Wo  = (const float*)d_in[5];
  const float* bo  = (const float*)d_in[6];
  const float* qd  = (const float*)d_in[7];
  const float* qu  = (const float*)d_in[8];
  const float* kd  = (const float*)d_in[9];
  const float* ku  = (const float*)d_in[10];
  const float* vd  = (const float*)d_in[11];
  const float* vu  = (const float*)d_in[12];
  const float* od  = (const float*)d_in[13];
  const float* ou  = (const float*)d_in[14];
  float* out = (float*)d_out;

  const size_t NHS = (size_t)16 * 4096 * 1280;  // 83,886,080
  const size_t NE  = (size_t)16 * 77 * 768;     // 946,176

  char* ws = (char*)d_ws;
  size_t off = 0;
  auto alloc = [&](size_t bytes) -> void* {
    void* p = ws + off;
    off += (bytes + 255) & ~(size_t)255;
    return p;
  };
  _Float16* hs16  = (_Float16*)alloc(NHS * 2);  // reused for attn_out
  _Float16* ehs16 = (_Float16*)alloc(NE * 2);
  _Float16* q16   = (_Float16*)alloc(NHS * 2);
  _Float16* k16   = (_Float16*)alloc((size_t)1232 * 1280 * 2);
  _Float16* v16   = (_Float16*)alloc((size_t)1232 * 1280 * 2);
  _Float16* wq16  = (_Float16*)alloc((size_t)1280 * 1280 * 2);  // [N,K]
  _Float16* wk16  = (_Float16*)alloc((size_t)1280 * 768 * 2);   // [N,K]
  _Float16* wv16  = (_Float16*)alloc((size_t)1280 * 768 * 2);   // [N,K]
  _Float16* wo16  = (_Float16*)alloc((size_t)1280 * 1280 * 2);  // [N,K]

  // 1) activations -> f16
  cvt_f32_f16<<<dim3((unsigned)(NHS / 4 / 256)), 256, 0, stream>>>(hs, hs16, (int)(NHS / 4));
  cvt_f32_f16<<<dim3((unsigned)(NE / 4 / 256)), 256, 0, stream>>>(ehs, ehs16, (int)(NE / 4));

  // 2) fold LoRA into weights, emit transposed [N,K] (LORA_SCALE == 1)
  fuse_lora_wT<<<dim3(5, 1280), 256, 0, stream>>>(Wq, qd, qu, wq16, 1280, 1280);
  fuse_lora_wT<<<dim3(5, 768),  256, 0, stream>>>(Wk, kd, ku, wk16, 1280, 768);
  fuse_lora_wT<<<dim3(5, 768),  256, 0, stream>>>(Wv, vd, vu, wv16, 1280, 768);
  fuse_lora_wT<<<dim3(5, 1280), 256, 0, stream>>>(Wo, od, ou, wo16, 1280, 1280);

  // 3) projections
  gemm_f16t<false><<<dim3(10, 512), 256, 0, stream>>>(hs16, wq16, q16, nullptr, 65536, 1280, 1280);
  gemm_f16t<false><<<dim3(10, 10),  256, 0, stream>>>(ehs16, wk16, k16, nullptr, 1232, 1280, 768);
  gemm_f16t<false><<<dim3(10, 10),  256, 0, stream>>>(ehs16, wv16, v16, nullptr, 1232, 1280, 768);

  // 4) attention (writes f16 attn_out into hs16 buffer; hs16 no longer needed)
  attn_kernel<<<dim3(32, 320), 256, 0, stream>>>(q16, k16, v16, hs16);

  // 5) output projection + bias -> f32 d_out
  gemm_f16t<true><<<dim3(10, 512), 256, 0, stream>>>(hs16, wo16, out, bo, 65536, 1280, 1280);
}
